// Net_4509715660893
// MI455X (gfx1250) — matrix operand order repaired and checked by reading, not compiled, not run on hardware
//
#include <hip/hip_runtime.h>

typedef __attribute__((ext_vector_type(4)))  float  v4f;
typedef __attribute__((ext_vector_type(8)))  float  v8f;
typedef __attribute__((ext_vector_type(4)))  __bf16 v4bf;
typedef __attribute__((ext_vector_type(8)))  __bf16 v8bf;
typedef __attribute__((ext_vector_type(16))) __bf16 v16bf;

#define GB 16
#define GN 2048
#define GF 128
#define GC 8

// ---------------- f32 -> bf16 conversion ----------------
__global__ void cvt_f32_bf16(const float* __restrict__ s, __bf16* __restrict__ d, long long n) {
    long long i = ((long long)blockIdx.x * blockDim.x + threadIdx.x) * 4;
    long long stride = (long long)gridDim.x * blockDim.x * 4;
    for (; i < n; i += stride) {
        v4f v = *(const v4f*)(s + i);
        v4bf o;
        o.x = (__bf16)v.x; o.y = (__bf16)v.y; o.z = (__bf16)v.z; o.w = (__bf16)v.w;
        *(v4bf*)(d + i) = o;
    }
}

// CDNA5 async global->LDS DMA (16B per lane), tracked by ASYNCcnt
__device__ __forceinline__ void async_copy_b128(unsigned lds_off, const void* gaddr) {
    asm volatile("global_load_async_to_lds_b128 %0, %1, off"
                 :: "v"(lds_off), "v"(gaddr) : "memory");
}
__device__ __forceinline__ unsigned lds_offset(const void* p) {
    // generic pointer into LDS: low 32 bits are the LDS byte address (ISA 10.2 aperture rule)
    return (unsigned)(unsigned long long)p;
}

// ---------------- Chebyshev step GEMM: Out = bf16(alpha * A@Bm + beta * Tpp) ----------------
// A: [GN x GN] bf16 (per batch), Bm: [GN x GF] bf16, Tpp: [GN x GF] bf16 (nullable)
#define BM 128
#define BK 64
#define LDA 72
#define LDB 136

__launch_bounds__(256)
__global__ void cheb_gemm(const __bf16* __restrict__ A,
                          const __bf16* __restrict__ Bm,
                          const __bf16* __restrict__ Tpp,
                          float alpha, float beta,
                          __bf16* __restrict__ Out)
{
    __shared__ __bf16 As[2][BM * LDA];   // 2 x 18 KB
    __shared__ __bf16 Bs[2][BK * LDB];   // 2 x 17 KB

    const int tid  = threadIdx.x;
    const int lane = tid & 31;
    const int wid  = tid >> 5;
    const int hlf  = (lane >> 4) & 1;
    const int l16  = lane & 15;
    const int wm   = (wid & 3) * 32;   // wave M offset within 128
    const int wn   = (wid >> 2) * 64;  // wave N offset within 128

    const int batch = blockIdx.y;
    const int mblk  = blockIdx.x;

    const __bf16* Ab = A  + (size_t)batch * GN * GN + (size_t)mblk * BM * GN;
    const __bf16* Bb = Bm + (size_t)batch * GN * GF;

    // per-thread DMA coordinates (16B chunks)
    const int ar = tid >> 3;           // 0..31 (A tile row base)
    const int ac = (tid & 7) * 8;      // A tile col (bf16)
    const int br = tid >> 4;           // 0..15 (B tile row base)
    const int bc = (tid & 15) * 8;     // B tile col (bf16)

    auto issue_stage = [&](int buf, int kt) {
#pragma unroll
        for (int p = 0; p < 4; ++p)
            async_copy_b128(lds_offset(&As[buf][(ar + p * 32) * LDA + ac]),
                            Ab + (size_t)(ar + p * 32) * GN + kt + ac);
#pragma unroll
        for (int p = 0; p < 4; ++p)
            async_copy_b128(lds_offset(&Bs[buf][(br + p * 16) * LDB + bc]),
                            Bb + (size_t)(kt + br + p * 16) * GF + bc);
    };

    v8f acc[2][4] = {};

    const int NSTAGE = GN / BK;        // 32
    issue_stage(0, 0);

    for (int s = 0; s < NSTAGE; ++s) {
        const int cur = s & 1;
        if (s + 1 < NSTAGE) {
            issue_stage(cur ^ 1, (s + 1) * BK);
            // 8 newer DMAs outstanding; <=8 means stage `s` has fully landed (in-order per wave)
            asm volatile("s_wait_asynccnt 8" ::: "memory");
        } else {
            asm volatile("s_wait_asynccnt 0" ::: "memory");
        }
        __syncthreads();   // publish all waves' LDS writes

        const __bf16* Asb = As[cur];
        const __bf16* Bsb = Bs[cur];
#pragma unroll
        for (int kc = 0; kc < BK; kc += 32) {
            v16bf afrag[2], bfrag[4];
            // A 16x32 fragment: lane holds row M=l16; elems 0..7 at K=kc+8*hlf,
            // elems 8..15 at K=kc+16+8*hlf (ISA 7.12.2 16-bit A layout)
#pragma unroll
            for (int i = 0; i < 2; ++i) {
                const __bf16* ap = Asb + (wm + 16 * i + l16) * LDA + kc + 8 * hlf;
                v8bf lo = *(const v8bf*)ap;
                v8bf hi = *(const v8bf*)(ap + 16);
                afrag[i] = __builtin_shufflevector(lo, hi, 0,1,2,3,4,5,6,7,8,9,10,11,12,13,14,15);
            }
            // B 32x16 fragment: lane holds column N=l16; elems 0..7 at K=kc+8*hlf,
            // elems 8..15 at K=kc+16+8*hlf (same K order as A)
#pragma unroll
            for (int j = 0; j < 4; ++j) {
                const __bf16* bp = Bsb + (kc + 8 * hlf) * LDB + wn + 16 * j + l16;
#pragma unroll
                for (int e = 0; e < 8; ++e) {
                    bfrag[j][e] = bp[e * LDB];
                    bfrag[j][8 + e] = bp[(16 + e) * LDB];
                }
            }
#pragma unroll
            for (int i = 0; i < 2; ++i)
#pragma unroll
                for (int j = 0; j < 4; ++j)
                    acc[i][j] = __builtin_amdgcn_wmma_f32_16x16x32_bf16(
                        false, afrag[i], false, bfrag[j],
                        (short)0, acc[i][j], false, false);
        }
        __syncthreads();   // buffer `cur` free for reuse two stages later
    }

    // ---- epilogue: alpha*acc + beta*Tpp, write bf16 ----
    const size_t obase = (size_t)batch * GN * GF + (size_t)mblk * BM * GF;
    __bf16* Ob = Out + obase;
    const __bf16* Pp = Tpp ? (Tpp + obase) : (const __bf16*)nullptr;

#pragma unroll
    for (int i = 0; i < 2; ++i) {
#pragma unroll
        for (int j = 0; j < 4; ++j) {
            const int col = wn + 16 * j + l16;
#pragma unroll
            for (int v = 0; v < 8; ++v) {
                const int row = wm + 16 * i + 8 * hlf + v;   // C/D layout: M = v + 8*half
                const size_t idx = (size_t)row * GF + col;
                float val = alpha * acc[i][j][v];
                if (Pp) val += beta * (float)Pp[idx];
                Ob[idx] = (__bf16)val;
            }
        }
    }
}

// ---------------- zero pooled ----------------
__global__ void zero_pooled(float* __restrict__ p) {
    int i = threadIdx.x;
    if (i < GB * GC) p[i] = 0.0f;
}

// ---------------- combine: h = relu(sum_k T_k @ W_k + b); pooled[b] += sum_n h ----------------
__launch_bounds__(256)
__global__ void combine_pool(const __bf16* __restrict__ t0, const __bf16* __restrict__ t1,
                             const __bf16* __restrict__ t2, const __bf16* __restrict__ t3,
                             const float* __restrict__ cw, const float* __restrict__ cb,
                             float* __restrict__ pooled)
{
    __shared__ float wsm[4 * GF * GC];   // 16 KB
    __shared__ float red[256 * GC];      // 8 KB
    const int tid = threadIdx.x;
    for (int i = tid; i < 4 * GF * GC; i += 256) wsm[i] = cw[i];
    __syncthreads();

    const int b = blockIdx.y;
    const int n = blockIdx.x * 256 + tid;
    const size_t base = (size_t)b * GN * GF + (size_t)n * GF;
    const __bf16* Ts[4] = { t0 + base, t1 + base, t2 + base, t3 + base };

    float h[GC];
#pragma unroll
    for (int c = 0; c < GC; ++c) h[c] = cb[c];

    for (int k = 0; k < 4; ++k) {
        const __bf16* tp = Ts[k];
        const float* wk = wsm + k * GF * GC;
        for (int f = 0; f < GF; ++f) {
            const float tv = (float)tp[f];
#pragma unroll
            for (int c = 0; c < GC; ++c) h[c] += tv * wk[f * GC + c];
        }
    }
#pragma unroll
    for (int c = 0; c < GC; ++c) red[tid * GC + c] = fmaxf(h[c], 0.0f);
    __syncthreads();

    for (int s = 128; s > 0; s >>= 1) {
        if (tid < s) {
#pragma unroll
            for (int c = 0; c < GC; ++c) red[tid * GC + c] += red[(tid + s) * GC + c];
        }
        __syncthreads();
    }
    if (tid < GC) atomicAdd(&pooled[b * GC + tid], red[tid]);
}

// ---------------- MLP head: pooled[B,8] -> Dense(32,relu) -> Dense(16,relu) -> Dense(1) ----------------
__global__ void head_mlp(const float* __restrict__ pooled,
                         const float* __restrict__ w1, const float* __restrict__ b1,
                         const float* __restrict__ w2, const float* __restrict__ b2,
                         const float* __restrict__ w3, const float* __restrict__ b3,
                         float* __restrict__ out)
{
    const int b = threadIdx.x;
    if (b >= GB) return;
    float p[GC];
#pragma unroll
    for (int c = 0; c < GC; ++c) p[c] = pooled[b * GC + c];
    float h1[32];
    for (int j = 0; j < 32; ++j) {
        float s = b1[j];
#pragma unroll
        for (int c = 0; c < GC; ++c) s += p[c] * w1[c * 32 + j];
        h1[j] = fmaxf(s, 0.0f);
    }
    float h2[16];
    for (int j = 0; j < 16; ++j) {
        float s = b2[j];
        for (int i = 0; i < 32; ++i) s += h1[i] * w2[i * 16 + j];
        h2[j] = fmaxf(s, 0.0f);
    }
    float o = b3[0];
    for (int i = 0; i < 16; ++i) o += h2[i] * w3[i];
    out[b] = o;
}

extern "C" void kernel_launch(void* const* d_in, const int* in_sizes, int n_in,
                              void* d_out, int out_size, void* d_ws, size_t ws_size,
                              hipStream_t stream) {
    const float* x  = (const float*)d_in[0];
    const float* a  = (const float*)d_in[1];
    const float* cw = (const float*)d_in[2];
    const float* cb = (const float*)d_in[3];
    const float* w1 = (const float*)d_in[4];
    const float* b1 = (const float*)d_in[5];
    const float* w2 = (const float*)d_in[6];
    const float* b2 = (const float*)d_in[7];
    const float* w3 = (const float*)d_in[8];
    const float* b3 = (const float*)d_in[9];
    float* out = (float*)d_out;

    // workspace layout (bytes)
    char* ws = (char*)d_ws;
    __bf16* Abf  = (__bf16*)(ws + 0);                         // 16*2048*2048*2 = 134217728
    __bf16* t0bf = (__bf16*)(ws + 134217728LL);               // 16*2048*128*2  = 8388608
    __bf16* t1bf = (__bf16*)(ws + 142606336LL);
    __bf16* t2bf = (__bf16*)(ws + 150994944LL);
    __bf16* t3bf = (__bf16*)(ws + 159383552LL);
    float*  pooled = (float*)(ws + 167772160LL);              // 16*8*4 = 512

    const long long nA = (long long)GB * GN * GN;
    const long long nX = (long long)GB * GN * GF;
    cvt_f32_bf16<<<8192, 256, 0, stream>>>(a, Abf, nA);
    cvt_f32_bf16<<<1024, 256, 0, stream>>>(x, t0bf, nX);

    dim3 gg(GN / BM, GB);   // 16 x 16 blocks
    cheb_gemm<<<gg, 256, 0, stream>>>(Abf, t0bf, nullptr, 1.0f,  0.0f, t1bf);
    cheb_gemm<<<gg, 256, 0, stream>>>(Abf, t1bf, t0bf,    2.0f, -1.0f, t2bf);
    cheb_gemm<<<gg, 256, 0, stream>>>(Abf, t2bf, t1bf,    2.0f, -1.0f, t3bf);

    zero_pooled<<<1, 256, 0, stream>>>(pooled);
    dim3 gc(GN / 256, GB);
    combine_pool<<<gc, 256, 0, stream>>>(t0bf, t1bf, t2bf, t3bf, cw, cb, pooled);
    head_mlp<<<1, 32, 0, stream>>>(pooled, w1, b1, w2, b2, w3, b3, out);
}
